// MAKT_90039694394081
// MI455X (gfx1250) — matrix operand order repaired and checked by reading, not compiled, not run on hardware
//
#include <hip/hip_runtime.h>
#include <math.h>

// ---------------------------------------------------------------------------
// MI455X (gfx1250, wave32, WMMA + TDM) implementation of the AKT-style
// transformer stack. All GEMMs and both attention matmuls use
// v_wmma_f32_16x16x32_f16 staged through LDS with fragment-contiguous
// (transposed) B layout; GEMM B tiles are DMA'd by the Tensor Data Mover
// (6-arg clang-23 builtin form, confirmed by the round-2 diagnostic).
// ---------------------------------------------------------------------------

constexpr int Bb  = 8;
constexpr int Ss  = 1024;
constexpr int Dd  = 512;
constexpr int Hh  = 8;
constexpr int Dff = 2048;
constexpr int NBv = 2;
constexpr int DK  = Dd / Hh;   // 64

typedef __attribute__((ext_vector_type(16))) _Float16 v16h;
typedef __attribute__((ext_vector_type(8)))  float    v8f;
typedef __attribute__((ext_vector_type(4)))  unsigned int u32x4;
typedef __attribute__((ext_vector_type(8)))  int      i32x8;
typedef __attribute__((ext_vector_type(4)))  int      i32x4;

#if defined(__has_builtin)
#if __has_builtin(__builtin_amdgcn_tensor_load_to_lds) && \
    __has_builtin(__builtin_amdgcn_s_wait_tensorcnt)
#define USE_TDM 1
#endif
#endif
#ifndef USE_TDM
#define USE_TDM 0
#endif

__device__ __forceinline__ v8f wmma16(v16h a, v16h b, v8f c) {
  // (neg_a, A, neg_b, B, c_mod, C, reuse_a, reuse_b)
  return __builtin_amdgcn_wmma_f32_16x16x32_f16(false, a, false, b,
                                                (short)0, c, false, false);
}
__device__ __forceinline__ v8f zero8() {
  v8f z = {0.f, 0.f, 0.f, 0.f, 0.f, 0.f, 0.f, 0.f};
  return z;
}

// wave32 reductions / scan (gfx1250 is wave32 only)
__device__ __forceinline__ float wave_max(float v) {
  for (int o = 16; o > 0; o >>= 1) v = fmaxf(v, __shfl_xor(v, o, 32));
  return v;
}
__device__ __forceinline__ float wave_sum(float v) {
  for (int o = 16; o > 0; o >>= 1) v += __shfl_xor(v, o, 32);
  return v;
}
__device__ __forceinline__ float wave_scan_incl(float v, int lane) {
  for (int o = 1; o < 32; o <<= 1) {
    float t = __shfl_up(v, o, 32);
    if (lane >= o) v += t;
  }
  return v;
}

// ---------------------------------------------------------------------------
// GEMM: C[M,N] = epilogue(A[M,K] @ W[K,N] + bias [+ res], relu?)
// 256 threads = 8 waves; block tile 128x64, wave tile 32x32, K-step 32.
// A tile: coalesced global loads + f16 convert into LDS (row-major, frag-
// contiguous). B tile: TDM DMA (f32) into LDS, then LDS->LDS transpose+cvt
// into Bst[N][K] so every B fragment is two contiguous ds_load_b128.
// ---------------------------------------------------------------------------
constexpr int BM = 128, BN = 64, BK = 32;

__global__ __launch_bounds__(256) void gemm_kernel(
    const float* __restrict__ A, const float* __restrict__ W,
    const float* __restrict__ bias, const float* __restrict__ res,
    float* __restrict__ C, int M, int K, int N, int relu)
{
  __shared__ alignas(16) _Float16 As[BM][BK + 8];   // 10 KB
  __shared__ alignas(16) _Float16 Bst[BN][BK + 8];  //  5 KB, transposed B
#if USE_TDM
  __shared__ alignas(16) float    Bs32[BK][BN];     //  8 KB, TDM landing pad
#endif

  int tid  = threadIdx.x;
  int lane = tid & 31, wid = tid >> 5;
  int bm = blockIdx.y * BM, bn = blockIdx.x * BN;
  int wm = (wid >> 1) * 32, wn = (wid & 1) * 32;
  int n16  = lane & 15;
  int mo   = (lane >> 4) << 3;   // C-frag row offset (M=r vs M=r+8)
  int ksel = mo;                  // unused; A frag K offset is koff
  int koff = (lane >> 4) << 4;    // B-frag K offset per lane half

  v8f acc[2][2];
  acc[0][0] = zero8(); acc[0][1] = zero8();
  acc[1][0] = zero8(); acc[1][1] = zero8();

  int ar = tid >> 1, acb = (tid & 1) * 16;  // A tile: 128 rows x 32 cols

#pragma unroll 1
  for (int k0 = 0; k0 < K; k0 += BK) {
#if USE_TDM
    if (wid == 0) {
      // Tensor DMA descriptor (ISA ch.8): 2D f32 tensor [K x N], tile BK x BN
      // starting at (k0, bn). global_addr = tile start; dim0 stride = N.
      unsigned long long ga =
          (unsigned long long)(W + (size_t)k0 * N + bn);
      unsigned int lds_off = (unsigned int)(size_t)&Bs32[0][0];
      u32x4 g0;
      g0[0] = 1u;                                   // count=1 (user D#)
      g0[1] = lds_off;                              // lds_addr
      g0[2] = (unsigned int)ga;                     // global_addr[31:0]
      g0[3] = (unsigned int)((ga >> 32) & 0x01FFFFFFu) | (2u << 30); // +type=2
      i32x8 g1;
      g1[0] = (int)(2u << 16);                      // data_size=2 (4 bytes)
      g1[1] = (int)((unsigned)N << 16);             // tensor_dim0 lo16
      g1[2] = (int)(((unsigned)N >> 16) | ((unsigned)K << 16)); // d0hi|d1lo
      g1[3] = (int)(((unsigned)K >> 16) | ((unsigned)BN << 16));// d1hi|tile0
      g1[4] = (int)BK;                              // tile_dim1 (tile2=0)
      g1[5] = (int)N;                               // dim0_stride[31:0]
      g1[6] = 0;
      g1[7] = 0;
      i32x4 gz4 = {0, 0, 0, 0};
      i32x8 gz8 = {0, 0, 0, 0, 0, 0, 0, 0};
      __builtin_amdgcn_tensor_load_to_lds(g0, g1, gz4, gz4, gz8, 0);
    }
#endif
    // A tile: coalesced load + convert (half row per thread)
    {
      const float* ap = A + (size_t)(bm + ar) * K + k0 + acb;
#pragma unroll
      for (int i = 0; i < 16; ++i) As[ar][acb + i] = (_Float16)ap[i];
      if (k0 + BK < K) __builtin_prefetch(ap + BK, 0, 1);  // global_prefetch_b8
    }
#if USE_TDM
    if (wid == 0) __builtin_amdgcn_s_wait_tensorcnt(0);
    __syncthreads();
    // LDS->LDS transpose + f32->f16 convert into fragment-contiguous layout
#pragma unroll
    for (int i = 0; i < 8; ++i) {
      int e = tid * 8 + i;
      int r = e >> 6;          // K row within tile
      int c = e & (BN - 1);    // N column
      Bst[c][r] = (_Float16)Bs32[r][c];
    }
#else
    // Fallback: direct transposed store global->LDS
    {
      int br = tid >> 3, bcb = (tid & 7) * 8;
      const float* wp = W + (size_t)(k0 + br) * N + bn + bcb;
#pragma unroll
      for (int i = 0; i < 8; ++i) Bst[bcb + i][br] = (_Float16)wp[i];
    }
#endif
    __syncthreads();

#pragma unroll
    for (int mi = 0; mi < 2; ++mi) {
      v16h a;
      int arow = wm + mi * 16 + (lane & 15);
#pragma unroll
      for (int vv = 0; vv < 8; ++vv) {
        int kb = 2 * vv + koff;
        a[2 * vv]     = As[arow][kb];
        a[2 * vv + 1] = As[arow][kb + 1];
      }
#pragma unroll
      for (int ni = 0; ni < 2; ++ni) {
        v16h bf;
        int bcol = wn + ni * 16 + n16;
#pragma unroll
        for (int vv = 0; vv < 8; ++vv) {        // contiguous: 2x ds_load_b128
          bf[2 * vv]     = Bst[bcol][koff + 2 * vv];
          bf[2 * vv + 1] = Bst[bcol][koff + 2 * vv + 1];
        }
        acc[mi][ni] = wmma16(a, bf, acc[mi][ni]);
      }
    }
    __syncthreads();
  }

#pragma unroll
  for (int mi = 0; mi < 2; ++mi)
#pragma unroll
    for (int ni = 0; ni < 2; ++ni)
#pragma unroll
      for (int r = 0; r < 8; ++r) {
        int row = bm + wm + mi * 16 + mo + r;
        int col = bn + wn + ni * 16 + n16;
        float x = acc[mi][ni][r];
        if (bias) x += bias[col];
        if (relu) x = fmaxf(x, 0.f);
        if (res)  x += res[(size_t)row * N + col];
        C[(size_t)row * N + col] = x;
      }
}

// ---------------------------------------------------------------------------
// Attention: one wave per (b, h, 16-query-row strip). Full 16x1024 score
// strip in LDS (f16). Faithful to reference: softmax over unmasked logits,
// mask, cumsum, distance-decay effect, re-masked softmax, optional zero_pad.
// Q@K^T and P@V via WMMA; V chunks staged transposed in LDS so B fragments
// are contiguous ds_load_b128 instead of D-strided global gathers.
// ---------------------------------------------------------------------------
__global__ __launch_bounds__(32) void attn_kernel(
    const float* __restrict__ qm, const float* __restrict__ km,
    const float* __restrict__ vm, const float* __restrict__ gam,
    float* __restrict__ om, int mask_k, int zero_pad)
{
  __shared__ alignas(16) _Float16 sc[16][Ss + 8];
  __shared__ alignas(16) _Float16 qt[16][DK];
  __shared__ alignas(16) _Float16 vt[DK][32 + 8];  // transposed V chunk

  int lane = threadIdx.x;
  int q0 = blockIdx.x * 16;
  int h  = blockIdx.y;
  int b  = blockIdx.z;
  const float scale = 0.125f;  // 1/sqrt(64)

  for (int t = lane; t < 16 * DK; t += 32) {
    int r = t >> 6, c = t & (DK - 1);
    qt[r][c] = (_Float16)qm[((size_t)(b * Ss + q0 + r)) * Dd + h * DK + c];
  }
  __syncthreads();

  int n16  = lane & 15;
  int mo   = (lane >> 4) << 3;
  int ksel = mo;
  int koff = (lane >> 4) << 4;

  // ---- scores = (Q @ K^T) * scale ----
#pragma unroll 1
  for (int jt = 0; jt < Ss / 16; ++jt) {
    v8f c = zero8();
#pragma unroll
    for (int kc = 0; kc < DK; kc += 32) {
      v16h a, bf;
#pragma unroll
      for (int vv = 0; vv < 8; ++vv) {
        int kb = 2 * vv + koff;
        a[2 * vv]     = qt[lane & 15][kc + kb];
        a[2 * vv + 1] = qt[lane & 15][kc + kb + 1];
      }
      const float* kp =
          km + ((size_t)(b * Ss + jt * 16 + n16)) * Dd + h * DK + kc + koff;
#pragma unroll
      for (int vv = 0; vv < 8; ++vv) {   // 16 contiguous f32: b128 loads
        bf[2 * vv]     = (_Float16)kp[2 * vv];
        bf[2 * vv + 1] = (_Float16)kp[2 * vv + 1];
      }
      c = wmma16(a, bf, c);
    }
#pragma unroll
    for (int r = 0; r < 8; ++r)
      sc[mo + r][jt * 16 + n16] = (_Float16)(c[r] * scale);
  }
  __syncthreads();

  float gamma = -log1pf(__expf(gam[h]));  // -softplus(gam)

  // ---- per-row softmax / cumsum / distance effect / re-softmax ----
#pragma unroll 1
  for (int r = 0; r < 16; ++r) {
    int gi  = q0 + r;
    int lim = gi + mask_k;  // mask[i][j] <=> j < i + mask_k
    float mx = -3.0e38f;
#pragma unroll 1
    for (int t = 0; t < 32; ++t) mx = fmaxf(mx, (float)sc[r][lane * 32 + t]);
    mx = wave_max(mx);
    float sum = 0.f;
#pragma unroll 1
    for (int t = 0; t < 32; ++t) sum += __expf((float)sc[r][lane * 32 + t] - mx);
    sum = wave_sum(sum);
    float inv = 1.f / sum;

    float csum = 0.f;  // masked prob chunk sum (lane owns 32 contiguous cols)
#pragma unroll 1
    for (int t = 0; t < 32; ++t) {
      int j = lane * 32 + t;
      if (j < lim) csum += __expf((float)sc[r][j] - mx) * inv;
    }
    float incl = wave_scan_incl(csum, lane);
    float tot  = __shfl(incl, 31, 32);   // disttot
    float run  = incl - csum;            // exclusive prefix
#pragma unroll 1
    for (int t = 0; t < 32; ++t) {
      int j = lane * 32 + t;
      float sraw = (float)sc[r][j];
      float p = (j < lim) ? __expf(sraw - mx) * inv : 0.f;
      run += p;                          // inclusive cumsum
      float posv = fabsf((float)(gi - j));
      float dist = sqrtf(fmaxf((tot - run) * posv, 0.f));
      float eff  = fminf(fmaxf(__expf(dist * gamma), 1e-5f), 1e5f);
      sc[r][j] = (_Float16)((j < lim) ? sraw * eff : -30000.f);
    }

    float mx2 = -3.0e38f;
#pragma unroll 1
    for (int t = 0; t < 32; ++t) mx2 = fmaxf(mx2, (float)sc[r][lane * 32 + t]);
    mx2 = wave_max(mx2);
    float sum2 = 0.f;
#pragma unroll 1
    for (int t = 0; t < 32; ++t) sum2 += __expf((float)sc[r][lane * 32 + t] - mx2);
    sum2 = wave_sum(sum2);
    float inv2 = 1.f / sum2;
    int zp = (zero_pad && gi == 0) ? 1 : 0;
#pragma unroll 1
    for (int t = 0; t < 32; ++t) {
      int j = lane * 32 + t;
      float pv = zp ? 0.f : __expf((float)sc[r][j] - mx2) * inv2;
      sc[r][j] = (_Float16)pv;
    }
  }
  __syncthreads();

  // ---- out = P @ V ----
  v8f accs[4];
#pragma unroll
  for (int nf = 0; nf < 4; ++nf) accs[nf] = zero8();
#pragma unroll 1
  for (int c0 = 0; c0 < Ss; c0 += 32) {
    // stage V chunk transposed: lane owns row c0+lane (64 contiguous f32 in)
    {
      const float* vrow = vm + ((size_t)(b * Ss + c0 + lane)) * Dd + h * DK;
#pragma unroll 8
      for (int c = 0; c < DK; ++c) vt[c][lane] = (_Float16)vrow[c];
    }
    __syncthreads();  // single-wave WG: lowers to S_NOP, orders LDS
    v16h a;
#pragma unroll
    for (int vv = 0; vv < 8; ++vv) {
      int kb = 2 * vv + koff;
      a[2 * vv]     = sc[lane & 15][c0 + kb];
      a[2 * vv + 1] = sc[lane & 15][c0 + kb + 1];
    }
#pragma unroll
    for (int nf = 0; nf < 4; ++nf) {
      v16h bf;
      int bcol = nf * 16 + n16;
#pragma unroll
      for (int vv = 0; vv < 8; ++vv) {          // contiguous: 2x ds_load_b128
        bf[2 * vv]     = vt[bcol][koff + 2 * vv];
        bf[2 * vv + 1] = vt[bcol][koff + 2 * vv + 1];
      }
      accs[nf] = wmma16(a, bf, accs[nf]);
    }
    __syncthreads();  // vt reused next chunk
  }
#pragma unroll
  for (int nf = 0; nf < 4; ++nf)
#pragma unroll
    for (int r = 0; r < 8; ++r)
      om[((size_t)(b * Ss + q0 + mo + r)) * Dd + h * DK + nf * 16 + n16] =
          accs[nf][r];
}

// ---------------------------------------------------------------------------
// LayerNorm over last dim (D=512), one row per 128-thread block.
// ---------------------------------------------------------------------------
__global__ __launch_bounds__(128) void ln_kernel(
    const float* __restrict__ in, const float* __restrict__ g,
    const float* __restrict__ bt, float* __restrict__ out)
{
  __shared__ float red[128];
  int row = blockIdx.x, tid = threadIdx.x;
  const float* x = in + (size_t)row * Dd;

  float s = 0.f;
  for (int i = tid; i < Dd; i += 128) s += x[i];
  red[tid] = s;
  __syncthreads();
  for (int o = 64; o > 0; o >>= 1) { if (tid < o) red[tid] += red[tid + o]; __syncthreads(); }
  float mu = red[0] / (float)Dd;
  __syncthreads();

  float vs = 0.f;
  for (int i = tid; i < Dd; i += 128) { float d = x[i] - mu; vs += d * d; }
  red[tid] = vs;
  __syncthreads();
  for (int o = 64; o > 0; o >>= 1) { if (tid < o) red[tid] += red[tid + o]; __syncthreads(); }
  float rstd = rsqrtf(red[0] / (float)Dd + 1e-5f);
  __syncthreads();

  for (int i = tid; i < Dd; i += 128)
    out[(size_t)row * Dd + i] = (x[i] - mu) * rstd * g[i] + bt[i];
}

// ---------------------------------------------------------------------------
// Host orchestration
// ---------------------------------------------------------------------------
struct LayerParams {
  const float *gam, *kW, *kb, *l1W, *l1b, *l2W, *l2b;
  const float *ln1b, *ln1g, *ln2b, *ln2g, *oW, *ob, *vW, *vb;
};

// jax pytree flattens dicts in sorted-key order:
// gam,kW,kb,l1W,l1b,l2W,l2b,ln1b,ln1g,ln2b,ln2g,oW,ob,vW,vb (15 leaves)
static void getParams(void* const* d_in, int base, int layer, LayerParams& P) {
  auto f = [&](int i) { return (const float*)d_in[base + i]; };
  size_t L = (size_t)layer;
  P.gam  = f(0)  + L * Hh;
  P.kW   = f(1)  + L * Dd * Dd;
  P.kb   = f(2)  + L * Dd;
  P.l1W  = f(3)  + L * Dd * Dff;
  P.l1b  = f(4)  + L * Dff;
  P.l2W  = f(5)  + L * Dff * Dd;
  P.l2b  = f(6)  + L * Dd;
  P.ln1b = f(7)  + L * Dd;
  P.ln1g = f(8)  + L * Dd;
  P.ln2b = f(9)  + L * Dd;
  P.ln2g = f(10) + L * Dd;
  P.oW   = f(11) + L * Dd * Dd;
  P.ob   = f(12) + L * Dd;
  P.vW   = f(13) + L * Dd * Dd;
  P.vb   = f(14) + L * Dd;
}

static void run_gemm(const float* A, const float* W, const float* bias,
                     const float* res, float* C, int M, int K, int N, int relu,
                     hipStream_t s) {
  dim3 g(N / BN, M / BM), b(256);
  gemm_kernel<<<g, b, 0, s>>>(A, W, bias, res, C, M, K, N, relu);
}

static void tlayer(const LayerParams& P, int mask_k, const float* query,
                   const float* values, bool apply_pos, float* out,
                   float* qk, float* vbuf, float* at, float* t0, float* t1,
                   float* ff, hipStream_t s) {
  const int M = Bb * Ss;
  // kq_same=True and query==key_ at every call site -> one projection for q=k
  run_gemm(query,  P.kW, P.kb, nullptr, qk,   M, Dd, Dd, 0, s);
  run_gemm(values, P.vW, P.vb, nullptr, vbuf, M, Dd, Dd, 0, s);
  dim3 ga(Ss / 16, Hh, Bb);
  attn_kernel<<<ga, 32, 0, s>>>(qk, qk, vbuf, P.gam, at, mask_k,
                                (mask_k == 0) ? 1 : 0);
  run_gemm(at, P.oW, P.ob, /*res=*/query, t0, M, Dd, Dd, 0, s);
  if (apply_pos) {
    ln_kernel<<<M, 128, 0, s>>>(t0, P.ln1g, P.ln1b, t1);
    run_gemm(t1, P.l1W, P.l1b, nullptr, ff, M, Dd, Dff, 1, s);
    run_gemm(ff, P.l2W, P.l2b, /*res=*/t1, t0, M, Dff, Dd, 0, s);
    ln_kernel<<<M, 128, 0, s>>>(t0, P.ln2g, P.ln2b, out);
  } else {
    ln_kernel<<<M, 128, 0, s>>>(t0, P.ln1g, P.ln1b, out);
  }
}

extern "C" void kernel_launch(void* const* d_in, const int* in_sizes, int n_in,
                              void* d_out, int out_size, void* d_ws,
                              size_t ws_size, hipStream_t stream) {
  (void)in_sizes; (void)n_in; (void)out_size; (void)ws_size;
  const size_t N1  = (size_t)Bb * Ss * Dd;    // 4 M floats
  const size_t NFF = (size_t)Bb * Ss * Dff;   // 16 M floats
  float* ws = (float*)d_ws;
  float* qk = ws;
  float* vb = qk + N1;
  float* at = vb + N1;
  float* t0 = at + N1;
  float* t1 = t0 + N1;
  float* ff = t1 + N1;
  float* y  = ff + NFF;
  float* x  = y + N1;

  (void)hipMemcpyAsync(x, d_in[0], N1 * sizeof(float),
                       hipMemcpyDeviceToDevice, stream);
  (void)hipMemcpyAsync(y, d_in[1], N1 * sizeof(float),
                       hipMemcpyDeviceToDevice, stream);

  LayerParams P;
  for (int i = 0; i < NBv; ++i) {           // first stack on qa-embed
    getParams(d_in, 2, i, P);
    tlayer(P, 1, y, y, true, y, qk, vb, at, t0, t1, ff, stream);
  }
  for (int i = 0; i < 2 * NBv; ++i) {       // second stack on q-embed
    getParams(d_in, 17, i, P);
    if ((i & 1) == 0) {
      tlayer(P, 1, x, x, false, x, qk, vb, at, t0, t1, ff, stream);
    } else {
      float* o = (i == 2 * NBv - 1) ? (float*)d_out : x;
      tlayer(P, 0, x, y, true, o, qk, vb, at, t0, t1, ff, stream);
    }
  }
}
